// TokenLevelModel_3659312136208
// MI455X (gfx1250) — matrix operand order repaired and checked by reading, not compiled, not run on hardware
//
#include <hip/hip_runtime.h>

// ---------------------------------------------------------------------------
// TokenLevelModel for MI455X (gfx1250, wave32, WMMA)
//   Y[b,c,:] = any(b,c) * ( X[b,tok(b,c),:] @ W^T + bias )
// GEMM as bf16 hi/lo split (bf16x3) on v_wmma_f32_16x16x32_bf16:
//   near-fp32 accuracy at 8x the FLOP/instr of the fp32 WMMA path.
// Gather fused into GEMM A-row indexing (no Y materialization).
// Round 2: wave = 32Mx64N (2 M-tiles), block = 4 waves sharing one N-range
//   -> 24 WMMA : 24 b128 loads per k-step (was 12 : 20), B frags L1-shared.
// ---------------------------------------------------------------------------

typedef __attribute__((ext_vector_type(16))) __bf16 v16bf;
typedef __attribute__((ext_vector_type(8)))  float  v8f;

#define BATCH 128
#define SEQ   128
#define HID   768
#define K2    1536                 // 2*H, also N
#define MAXC  141
#define MROWS (BATCH * MAXC)       // 18048 output rows (= 141 * 128)
#define XROWS (BATCH * SEQ)        // 16384 X rows

static __device__ __forceinline__ unsigned short bf16_rn(float f) {
  unsigned u = __float_as_uint(f);
  u += 0x7FFFu + ((u >> 16) & 1u);          // round to nearest even
  return (unsigned short)(u >> 16);
}
static __device__ __forceinline__ float bf16_to_f(unsigned short h) {
  return __uint_as_float(((unsigned)h) << 16);
}

// --- prep: last-covering-token map --------------------------------------- //
__global__ void k_tokmap(const int* __restrict__ om,
                         int* __restrict__ tok, float* __restrict__ anyf) {
  int tid = blockIdx.x * blockDim.x + threadIdx.x;
  if (tid >= MROWS) return;
  int b = tid / MAXC, c = tid % MAXC;
  const int* p = om + (size_t)b * SEQ * 2;
  int last = 0, found = 0;
  for (int t = 0; t < SEQ; ++t) {
    int s0 = p[2 * t], e = p[2 * t + 1];
    if (c >= s0 && c < e) { last = t; found = 1; }   // later token overwrites
  }
  tok[tid]  = b * SEQ + (found ? last : 0);
  anyf[tid] = found ? 1.0f : 0.0f;
}

// --- prep: split X = concat(h0,h1) into bf16 hi/lo ----------------------- //
__global__ void k_convX(const float* __restrict__ h0, const float* __restrict__ h1,
                        unsigned short* __restrict__ Xh, unsigned short* __restrict__ Xl) {
  size_t idx = (size_t)blockIdx.x * blockDim.x + threadIdx.x;
  if (idx >= (size_t)XROWS * K2) return;
  size_t m = idx / K2; int k = (int)(idx % K2);
  float v = (k < HID) ? h0[m * HID + k] : h1[m * HID + (k - HID)];
  unsigned short hi = bf16_rn(v);
  Xh[idx] = hi;
  Xl[idx] = bf16_rn(v - bf16_to_f(hi));
}

// --- prep: split W into bf16 hi/lo --------------------------------------- //
__global__ void k_convW(const float* __restrict__ W,
                        unsigned short* __restrict__ Wh, unsigned short* __restrict__ Wl) {
  size_t idx = (size_t)blockIdx.x * blockDim.x + threadIdx.x;
  if (idx >= (size_t)K2 * K2) return;
  float v = W[idx];
  unsigned short hi = bf16_rn(v);
  Wh[idx] = hi;
  Wl[idx] = bf16_rn(v - bf16_to_f(hi));
}

// --- fused gather-GEMM ---------------------------------------------------- //
// Wave computes 32(M) x 64(N); block = 4 waves stacked in M (128 x 64),
// all sharing the same B (W) fragments. grid = (141, 24), block = 128.
union Frag { uint4 q[2]; v16bf v; };

__global__ void __launch_bounds__(128)
k_gemm(const unsigned short* __restrict__ Xh, const unsigned short* __restrict__ Xl,
       const unsigned short* __restrict__ Wh, const unsigned short* __restrict__ Wl,
       const float* __restrict__ bias, const int* __restrict__ tok,
       const float* __restrict__ anyf, float* __restrict__ out) {
  const int wave   = threadIdx.x >> 5;                 // 0..3
  const int lane   = threadIdx.x & 31;
  const int half   = lane >> 4;                        // lane group 0..1
  const int l15    = lane & 15;
  const int m_base = blockIdx.x * 128 + wave * 32;     // 32 rows per wave
  const int n_base = blockIdx.y * 64;                  // shared by all waves

  // A fragments: lane holds row M = l15 of each 16-row subtile -> gathered X rows
  const size_t xrow0 = (size_t)tok[m_base + l15];
  const size_t xrow1 = (size_t)tok[m_base + 16 + l15];
  const unsigned short* xh0 = Xh + xrow0 * K2;
  const unsigned short* xl0 = Xl + xrow0 * K2;
  const unsigned short* xh1 = Xh + xrow1 * K2;
  const unsigned short* xl1 = Xl + xrow1 * K2;

  v8f acc[2][4];
#pragma unroll
  for (int t = 0; t < 4; ++t) {
    float bv = bias[n_base + t * 16 + l15];            // bias depends on N only
#pragma unroll
    for (int r = 0; r < 8; ++r) { acc[0][t][r] = bv; acc[1][t][r] = bv; }
  }

  for (int k0 = 0; k0 < K2; k0 += 32) {
    // 16-bit A 16x32 layout: VGPR 0-3 hold K = 8*half+0..7, VGPR 4-7 hold K = 16+8*half+0..7
    Frag ah0, al0, ah1, al1;
    ah0.q[0] = *(const uint4*)(xh0 + k0 + 8 * half);
    ah0.q[1] = *(const uint4*)(xh0 + k0 + 16 + 8 * half);
    al0.q[0] = *(const uint4*)(xl0 + k0 + 8 * half);
    al0.q[1] = *(const uint4*)(xl0 + k0 + 16 + 8 * half);
    ah1.q[0] = *(const uint4*)(xh1 + k0 + 8 * half);
    ah1.q[1] = *(const uint4*)(xh1 + k0 + 16 + 8 * half);
    al1.q[0] = *(const uint4*)(xl1 + k0 + 8 * half);
    al1.q[1] = *(const uint4*)(xl1 + k0 + 16 + 8 * half);
#pragma unroll
    for (int t = 0; t < 4; ++t) {
      // B 32x16 layout: lane holds column N = l15, K = 8*half+0..7 and 16+8*half+0..7
      // B[k][n] = W[n][k]  (Y = X @ W^T) -> two 16-byte chunks of W row n, 32 bytes apart.
      const size_t n = (size_t)(n_base + t * 16 + l15);
      const unsigned short* wh = Wh + n * K2 + k0 + 8 * half;
      const unsigned short* wl = Wl + n * K2 + k0 + 8 * half;
      Frag bh, bl;
      bh.q[0] = *(const uint4*)(wh);
      bh.q[1] = *(const uint4*)(wh + 16);
      bl.q[0] = *(const uint4*)(wl);
      bl.q[1] = *(const uint4*)(wl + 16);
      // bf16x3: hi*hi + hi*lo + lo*hi (lo*lo below fp32 epsilon), both M-tiles
      acc[0][t] = __builtin_amdgcn_wmma_f32_16x16x32_bf16(false, ah0.v, false, bh.v,
                                                          (short)0, acc[0][t], false, false);
      acc[0][t] = __builtin_amdgcn_wmma_f32_16x16x32_bf16(false, ah0.v, false, bl.v,
                                                          (short)0, acc[0][t], false, false);
      acc[0][t] = __builtin_amdgcn_wmma_f32_16x16x32_bf16(false, al0.v, false, bh.v,
                                                          (short)0, acc[0][t], false, false);
      acc[1][t] = __builtin_amdgcn_wmma_f32_16x16x32_bf16(false, ah1.v, false, bh.v,
                                                          (short)0, acc[1][t], false, false);
      acc[1][t] = __builtin_amdgcn_wmma_f32_16x16x32_bf16(false, ah1.v, false, bl.v,
                                                          (short)0, acc[1][t], false, false);
      acc[1][t] = __builtin_amdgcn_wmma_f32_16x16x32_bf16(false, al1.v, false, bh.v,
                                                          (short)0, acc[1][t], false, false);
    }
  }

  // C/D f32 16x16 layout: VGPR r, lane -> (M = r + 8*half, N = l15)
#pragma unroll
  for (int g = 0; g < 2; ++g) {
#pragma unroll
    for (int t = 0; t < 4; ++t) {
#pragma unroll
      for (int r = 0; r < 8; ++r) {
        int row = m_base + g * 16 + r + 8 * half;
        out[(size_t)row * K2 + n_base + t * 16 + l15] = acc[g][t][r] * anyf[row];
      }
    }
  }
}

// --------------------------------------------------------------------------- //
extern "C" void kernel_launch(void* const* d_in, const int* in_sizes, int n_in,
                              void* d_out, int out_size, void* d_ws, size_t ws_size,
                              hipStream_t stream) {
  (void)in_sizes; (void)n_in; (void)out_size; (void)ws_size;
  const float* h0   = (const float*)d_in[0];
  const float* h1   = (const float*)d_in[1];
  const float* W    = (const float*)d_in[2];
  const float* bias = (const float*)d_in[3];
  const int*   om   = (const int*)d_in[4];
  float*       out  = (float*)d_out;

  char* ws = (char*)d_ws;
  size_t off = 0;
#define WS_ALLOC(ptrvar, type, count)                                          \
  type* ptrvar = (type*)(ws + off);                                            \
  off += (((size_t)(count) * sizeof(type)) + 255) & ~(size_t)255;

  WS_ALLOC(Xh,   unsigned short, (size_t)XROWS * K2)   // 50.3 MB
  WS_ALLOC(Xl,   unsigned short, (size_t)XROWS * K2)   // 50.3 MB
  WS_ALLOC(Wh,   unsigned short, (size_t)K2 * K2)      //  4.7 MB
  WS_ALLOC(Wl,   unsigned short, (size_t)K2 * K2)      //  4.7 MB
  WS_ALLOC(tok,  int,            MROWS)
  WS_ALLOC(anyf, float,          MROWS)
#undef WS_ALLOC

  {
    size_t n = (size_t)XROWS * K2;
    k_convX<<<(unsigned)((n + 255) / 256), 256, 0, stream>>>(h0, h1, Xh, Xl);
  }
  {
    size_t n = (size_t)K2 * K2;
    k_convW<<<(unsigned)((n + 255) / 256), 256, 0, stream>>>(W, Wh, Wl);
  }
  k_tokmap<<<(MROWS + 255) / 256, 256, 0, stream>>>(om, tok, anyf);

  dim3 grid(MROWS / 128, K2 / 64);   // (141, 24): 4 waves x 32 rows, 64 cols
  k_gemm<<<grid, 128, 0, stream>>>(Xh, Xl, Wh, Wl, bias, tok, anyf, out);
}
